// AASISTBackbone_66838281060957
// MI455X (gfx1250) — matrix operand order repaired and checked by reading, not compiled, not run on hardware
//
#include <hip/hip_runtime.h>
#include <hip/hip_bf16.h>

// ---------------------------------------------------------------------------
// AASIST backbone for MI455X (gfx1250, wave32).
// GEMMs (h@W, attn@Wh): f16-packed operands, double-buffered async-to-LDS
// staging (global_load_async_to_lds_b128 + s_wait_asynccnt), fragments via
// ds_load_b128, math on v_wmma_f32_16x16x32_f16 (2 tiles/wave, shared B frag).
// Convs/BN/softmax/ELU/pool: VALU kernels (tiny FLOPs, L2-resident).
// ---------------------------------------------------------------------------

typedef __attribute__((ext_vector_type(16))) _Float16 v16h;
typedef __attribute__((ext_vector_type(8)))  _Float16 v8h;
typedef __attribute__((ext_vector_type(8)))  float    v8f;
typedef __attribute__((ext_vector_type(4)))  int      v4i;

#define BN_EPS 1e-5f
#define LRELU_ALPHA 0.2f

// -------- async global->LDS 16B copy (gfx1250 path with fallback) ----------

#if defined(__HIP_DEVICE_COMPILE__) && __has_builtin(__builtin_amdgcn_global_load_async_to_lds_b128)
#define HAVE_ASYNC_LDS 1
#else
#define HAVE_ASYNC_LDS 0
#endif

__device__ __forceinline__ void cp16_g2s(void* lds, const void* g)
{
#if HAVE_ASYNC_LDS
    typedef __attribute__((address_space(1))) v4i gv4i;   // prints as __device__
    typedef __attribute__((address_space(3))) v4i lv4i;   // prints as __shared__
    __builtin_amdgcn_global_load_async_to_lds_b128((gv4i*)g, (lv4i*)lds, 0, 0);
#else
    *(v8h*)lds = *(const v8h*)g;          // global_load_b128 + ds_store_b128
#endif
}
__device__ __forceinline__ void cp_wait()
{
#if HAVE_ASYNC_LDS
    asm volatile("s_wait_asynccnt 0x0" ::: "memory");
#endif
}

// ------------------------- wave32 block reductions -------------------------

__device__ __forceinline__ float warp_max32(float v) {
    #pragma unroll
    for (int o = 16; o > 0; o >>= 1) v = fmaxf(v, __shfl_xor(v, o, 32));
    return v;
}
__device__ __forceinline__ float warp_sum32(float v) {
    #pragma unroll
    for (int o = 16; o > 0; o >>= 1) v += __shfl_xor(v, o, 32);
    return v;
}
__device__ __forceinline__ float block_max(float v, float* sm) {
    v = warp_max32(v);
    int lane = threadIdx.x & 31, w = threadIdx.x >> 5;
    int nw = (blockDim.x + 31) >> 5;
    if (lane == 0) sm[w] = v;
    __syncthreads();
    v = (threadIdx.x < (unsigned)nw) ? sm[threadIdx.x] : -3.0e38f;
    v = warp_max32(v);
    if (threadIdx.x == 0) sm[0] = v;
    __syncthreads();
    v = sm[0];
    __syncthreads();
    return v;
}
__device__ __forceinline__ float block_sum(float v, float* sm) {
    v = warp_sum32(v);
    int lane = threadIdx.x & 31, w = threadIdx.x >> 5;
    int nw = (blockDim.x + 31) >> 5;
    if (lane == 0) sm[w] = v;
    __syncthreads();
    v = (threadIdx.x < (unsigned)nw) ? sm[threadIdx.x] : 0.0f;
    v = warp_sum32(v);
    if (threadIdx.x == 0) sm[0] = v;
    __syncthreads();
    v = sm[0];
    __syncthreads();
    return v;
}

// ------------------------- conv1d (+BN, +skip, +ReLU) -----------------------

__global__ void conv1d_bn_kernel(const float* __restrict__ in, int B, int Cin, int T,
                                 const float* __restrict__ w, const float* __restrict__ bias,
                                 int Cout, int K, int pad,
                                 const float* __restrict__ g, const float* __restrict__ beta,
                                 const float* __restrict__ mu, const float* __restrict__ var,
                                 const float* __restrict__ skip, int do_relu,
                                 float* __restrict__ out)
{
    int idx = blockIdx.x * blockDim.x + threadIdx.x;
    int total = B * Cout * T;
    if (idx >= total) return;
    int t = idx % T;
    int o = (idx / T) % Cout;
    int b = idx / (T * Cout);

    const float* inb = in + (long)b * Cin * T;
    const float* wo  = w  + (long)o * Cin * K;
    __builtin_prefetch(inb + t, 0, 1);   // global_prefetch_b8

    float acc = bias[o];
    for (int i = 0; i < Cin; ++i) {
        const float* ir = inb + (long)i * T;
        const float* wr = wo + i * K;
        #pragma unroll 3
        for (int k = 0; k < K; ++k) {
            int tt = t + k - pad;
            if (tt >= 0 && tt < T) acc = fmaf(wr[k], ir[tt], acc);
        }
    }
    if (g) {
        float s = g[o] * rsqrtf(var[o] + BN_EPS);
        acc = (acc - mu[o]) * s + beta[o];
    }
    if (skip) acc += skip[(long)b * Cout * T + (long)o * T + t];
    if (do_relu) acc = fmaxf(acc, 0.0f);
    out[(long)b * Cout * T + (long)o * T + t] = acc;
}

// ------------------------- f32 (strided) -> f16 pack ------------------------
// dst[b][m][k] = (f16) src[b*sb + m*sm + k*sk]

__global__ void pack_f16_kernel(const float* __restrict__ src, long sm, long sk, long sb,
                                _Float16* __restrict__ dst, int M, int K, int NB)
{
    long idx = (long)blockIdx.x * blockDim.x + threadIdx.x;
    long total = (long)NB * M * K;
    if (idx >= total) return;
    int k = (int)(idx % K);
    long t = idx / K;
    int m = (int)(t % M);
    int b = (int)(t / M);
    dst[idx] = (_Float16)src[(long)b * sb + (long)m * sm + (long)k * sk];
}

// ------------------------- WMMA GEMM (f16 in, f32 out) ----------------------
// C[b] (MxN=64 row-major f32) = A[b] (MxK row-major f16) @ BT[b]^T
// (BT is NxK row-major f16). Block: 256 thr = 8 waves; 64x64 macro-tile.
// Wave (mHalf, nSub) computes two 16x16 tiles (shared B fragment, 2 WMMAs).
// Double-buffered LDS: copy for step k+1 issued after step-k barrier and
// overlapped with step-k ds_load+WMMA. Uniform copies: each thread moves one
// 16B A chunk + one 16B B chunk per step (no EXEC divergence).
// K % 32 == 0, M % 64 == 0, N == 64; grid sized exactly (uniform barriers).

__global__ void wmma_gemm_kernel(const _Float16* __restrict__ A, long sAb,
                                 const _Float16* __restrict__ BT, long sBb,
                                 float* __restrict__ C, long sCb,
                                 int M, int N, int Kdim)
{
    __shared__ _Float16 As[2][64][32];
    __shared__ _Float16 Bs[2][64][32];

    int tid  = threadIdx.x;
    int wave = tid >> 5;
    int lane = tid & 31;
    int mHalf = wave >> 2;         // 0..1 -> rows [mHalf*32, mHalf*32+32)
    int nSub  = wave & 3;          // 0..3 -> cols [nSub*16, nSub*16+16)
    int row0 = blockIdx.x * 64;

    const _Float16* Ab  = A  + (long)blockIdx.z * sAb;
    const _Float16* BTb = BT + (long)blockIdx.z * sBb;
    float*          Cb  = C  + (long)blockIdx.z * sCb;

    int hsel = lane >> 4;          // 0 | 1
    int lr   = lane & 15;
    int arow = mHalf * 32 + lr;    // tile0 row; tile1 = arow+16
    int bcol = nSub * 16 + lr;

    int rC = tid >> 2, pC = tid & 3;      // 64 rows x 4 16B pieces per tile

    const _Float16* gA = Ab  + (long)(row0 + rC) * Kdim + pC * 8;
    const _Float16* gB = BTb + (long)rC * Kdim + pC * 8;
    _Float16* lA0 = &As[0][rC][pC * 8];
    _Float16* lB0 = &Bs[0][rC][pC * 8];
    _Float16* lA1 = &As[1][rC][pC * 8];
    _Float16* lB1 = &Bs[1][rC][pC * 8];

    // prologue: stage k0 = 0 into buffer 0
    cp16_g2s(lA0, gA);
    cp16_g2s(lB0, gB);

    v8f acc0 = {}, acc1 = {};
    int nSteps = Kdim >> 5;
    for (int s = 0; s < nSteps; ++s) {
        int cur = s & 1;
        cp_wait();                 // own async copies for buffer `cur` done
        __syncthreads();           // all waves' copies done; prior reads of
                                   // buffer cur^1 complete -> safe to refill
        if (s + 1 < nSteps) {      // uniform: overlap next copy with compute
            long koff = (long)(s + 1) << 5;
            cp16_g2s(cur ? lA0 : lA1, gA + koff);
            cp16_g2s(cur ? lB0 : lB1, gB + koff);
        }

        v8h b_lo = *(const v8h*)&Bs[cur][bcol][hsel * 8];
        v8h b_hi = *(const v8h*)&Bs[cur][bcol][16 + hsel * 8];
        v16h b = __builtin_shufflevector(b_lo, b_hi, 0,1,2,3,4,5,6,7,8,9,10,11,12,13,14,15);

        v8h a0l = *(const v8h*)&As[cur][arow][hsel * 8];
        v8h a0h = *(const v8h*)&As[cur][arow][16 + hsel * 8];
        v16h a0 = __builtin_shufflevector(a0l, a0h, 0,1,2,3,4,5,6,7,8,9,10,11,12,13,14,15);
        acc0 = __builtin_amdgcn_wmma_f32_16x16x32_f16(
                   false, a0, false, b, (short)0, acc0, false, false);

        v8h a1l = *(const v8h*)&As[cur][arow + 16][hsel * 8];
        v8h a1h = *(const v8h*)&As[cur][arow + 16][16 + hsel * 8];
        v16h a1 = __builtin_shufflevector(a1l, a1h, 0,1,2,3,4,5,6,7,8,9,10,11,12,13,14,15);
        acc1 = __builtin_amdgcn_wmma_f32_16x16x32_f16(
                   false, a1, false, b, (short)0, acc1, false, false);
    }

    int tm = row0 + mHalf * 32;
    int tn = nSub * 16;
    #pragma unroll
    for (int r = 0; r < 8; ++r) {
        Cb[(long)(tm + r + hsel * 8) * N + (tn + lr)]      = acc0[r];
        Cb[(long)(tm + 16 + r + hsel * 8) * N + (tn + lr)] = acc1[r];
    }
}

// ------------------------- GAT pieces --------------------------------------

__global__ void gat_scores_kernel(const float* __restrict__ Wh, const float* __restrict__ a,
                                  int Fdim, int BN_total,
                                  float* __restrict__ f1, float* __restrict__ f2)
{
    int idx = blockIdx.x * blockDim.x + threadIdx.x;
    if (idx >= BN_total) return;
    const float* row = Wh + (long)idx * Fdim;
    float s1 = 0.f, s2 = 0.f;
    for (int f = 0; f < Fdim; ++f) {
        float h = row[f];
        s1 = fmaf(h, a[f], s1);
        s2 = fmaf(h, a[Fdim + f], s2);
    }
    f1[idx] = s1;
    f2[idx] = s2;
}

// attn[b][i][:] = softmax_j( lrelu(f1[i] + f2[j]) ), f16 output, block per (b,i)
__global__ void attn_row_kernel(const float* __restrict__ f1, const float* __restrict__ f2,
                                _Float16* __restrict__ attn, int Nn)
{
    __shared__ float sm[8];
    int b = blockIdx.x / Nn;
    int i = blockIdx.x % Nn;
    const float* f2b = f2 + (long)b * Nn;
    float fi = f1[(long)b * Nn + i];

    float lmax = -3.0e38f;
    for (int j = threadIdx.x; j < Nn; j += blockDim.x) {
        float e = fi + f2b[j];
        e = (e > 0.f) ? e : LRELU_ALPHA * e;
        lmax = fmaxf(lmax, e);
    }
    float gmax = block_max(lmax, sm);

    float lsum = 0.f;
    for (int j = threadIdx.x; j < Nn; j += blockDim.x) {
        float e = fi + f2b[j];
        e = (e > 0.f) ? e : LRELU_ALPHA * e;
        lsum += expf(e - gmax);
    }
    float gsum = block_sum(lsum, sm);
    float inv = 1.0f / gsum;

    _Float16* arow = attn + ((long)b * Nn + i) * Nn;
    for (int j = threadIdx.x; j < Nn; j += blockDim.x) {
        float e = fi + f2b[j];
        e = (e > 0.f) ? e : LRELU_ALPHA * e;
        arow[j] = (_Float16)(expf(e - gmax) * inv);
    }
}

// ELU -> f16 (GEMM A operand) and optionally f32 (pool input)
__global__ void elu_pack_kernel(const float* __restrict__ in,
                                _Float16* __restrict__ out_h,
                                float* __restrict__ out_f, int n)
{
    int idx = blockIdx.x * blockDim.x + threadIdx.x;
    if (idx >= n) return;
    float x = in[idx];
    float y = (x > 0.f) ? x : (expf(x) - 1.0f);
    out_h[idx] = (_Float16)y;
    if (out_f) out_f[idx] = y;
}

// ------------------------- attention pool ----------------------------------

__global__ void pool_logits_kernel(const float* __restrict__ h,
                                   const float* __restrict__ w1, const float* __restrict__ b1,
                                   const float* __restrict__ w2, const float* __restrict__ b2,
                                   float* __restrict__ logits, int BN_total, int Fdim, int Hd)
{
    int idx = blockIdx.x * blockDim.x + threadIdx.x;
    if (idx >= BN_total) return;
    const float* row = h + (long)idx * Fdim;
    float lg = b2[0];
    for (int j = 0; j < Hd; ++j) {
        float t = b1[j];
        for (int f = 0; f < Fdim; ++f) t = fmaf(row[f], w1[f * Hd + j], t);
        t = fmaxf(t, 0.f);
        lg = fmaf(t, w2[j], lg);
    }
    logits[idx] = lg;
}

__global__ void softmax_stats_kernel(const float* __restrict__ logits,
                                     float* __restrict__ stats, int Nn)
{
    __shared__ float sm[8];
    int b = blockIdx.x;
    const float* l = logits + (long)b * Nn;
    float lmax = -3.0e38f;
    for (int j = threadIdx.x; j < Nn; j += blockDim.x) lmax = fmaxf(lmax, l[j]);
    float gmax = block_max(lmax, sm);
    float lsum = 0.f;
    for (int j = threadIdx.x; j < Nn; j += blockDim.x) lsum += expf(l[j] - gmax);
    float gsum = block_sum(lsum, sm);
    if (threadIdx.x == 0) { stats[2 * b] = gmax; stats[2 * b + 1] = gsum; }
}

__global__ void pool_out_kernel(const float* __restrict__ h, const float* __restrict__ logits,
                                const float* __restrict__ stats, float* __restrict__ out,
                                int Nn, int Fdim)
{
    __shared__ float sm[8];
    int b = blockIdx.x / Fdim;
    int f = blockIdx.x % Fdim;
    float gmax = stats[2 * b];
    float inv  = 1.0f / stats[2 * b + 1];
    float acc = 0.f;
    for (int n = threadIdx.x; n < Nn; n += blockDim.x) {
        float wgt = expf(logits[(long)b * Nn + n] - gmax) * inv;
        acc = fmaf(wgt, h[((long)b * Nn + n) * Fdim + f], acc);
    }
    acc = block_sum(acc, sm);
    if (threadIdx.x == 0) out[(long)b * Fdim + f] = acc;
}

// ---------------------------------------------------------------------------

extern "C" void kernel_launch(void* const* d_in, const int* in_sizes, int n_in,
                              void* d_out, int out_size, void* d_ws, size_t ws_size,
                              hipStream_t stream)
{
    (void)in_sizes; (void)n_in; (void)out_size; (void)ws_size;
    const int B = 4, T = 512, Nn = 512;

    // ---- inputs (insertion-order flattening of setup_inputs dict) ----
    int ii = 0;
    const float* x    = (const float*)d_in[ii++];
    const float* ct_w = (const float*)d_in[ii++];
    const float* ct_b = (const float*)d_in[ii++];

    struct RB {
        const float *c1w, *c1b, *c2w, *c2b;
        const float *g1, *be1, *m1, *v1;
        const float *g2, *be2, *m2, *v2;
        const float *skw, *skb;
    } rb[3];
    for (int r = 0; r < 3; ++r) {
        rb[r].c1w = (const float*)d_in[ii++];
        rb[r].c1b = (const float*)d_in[ii++];
        rb[r].c2w = (const float*)d_in[ii++];
        rb[r].c2b = (const float*)d_in[ii++];
        rb[r].g1  = (const float*)d_in[ii++];
        rb[r].be1 = (const float*)d_in[ii++];
        rb[r].m1  = (const float*)d_in[ii++];
        rb[r].v1  = (const float*)d_in[ii++];
        rb[r].g2  = (const float*)d_in[ii++];
        rb[r].be2 = (const float*)d_in[ii++];
        rb[r].m2  = (const float*)d_in[ii++];
        rb[r].v2  = (const float*)d_in[ii++];
        if (r == 2) {
            rb[r].skw = (const float*)d_in[ii++];
            rb[r].skb = (const float*)d_in[ii++];
        } else { rb[r].skw = nullptr; rb[r].skb = nullptr; }
    }
    const float* W1  = (const float*)d_in[ii++];
    const float* a1  = (const float*)d_in[ii++];
    const float* W2  = (const float*)d_in[ii++];
    const float* a2  = (const float*)d_in[ii++];
    const float* pw1 = (const float*)d_in[ii++];
    const float* pb1 = (const float*)d_in[ii++];
    const float* pw2 = (const float*)d_in[ii++];
    const float* pb2 = (const float*)d_in[ii++];

    // ---- workspace layout ----
    float* ws = (float*)d_ws;
    const long CHBUF = (long)B * 128 * T;
    float* buf0   = ws;                 ws += CHBUF;
    float* buf1   = ws;                 ws += CHBUF;
    float* buf2   = ws;                 ws += CHBUF;
    float* skipb  = ws;                 ws += CHBUF;
    float* Whb    = ws;                 ws += (long)B * Nn * 64;
    float* f1b    = ws;                 ws += (long)B * Nn;
    float* f2b    = ws;                 ws += (long)B * Nn;
    float* hprime = ws;                 ws += (long)B * Nn * 64;
    float* hfin   = ws;                 ws += (long)B * Nn * 64;
    float* logits = ws;                 ws += (long)B * Nn;
    float* stats  = ws;                 ws += 2 * B;
    _Float16* a1pack = (_Float16*)ws;   ws += (long)B * Nn * 128 / 2;   // f16
    _Float16* btW    = (_Float16*)ws;   ws += (long)64 * 512 / 2;       // weights^T
    _Float16* btWh   = (_Float16*)ws;   ws += (long)B * 64 * Nn / 2;    // Wh^T
    _Float16* attnh  = (_Float16*)ws;   ws += (long)B * Nn * Nn / 2;    // attn f16
    _Float16* hgath  = (_Float16*)ws;   ws += (long)B * Nn * 64 / 2;    // elu f16

    const int TPB = 256;
    auto blocksFor = [](long n, int tpb) { return (int)((n + tpb - 1) / tpb); };

    // ---- front conv: (B,1,T) -> (B,64,T) ----
    conv1d_bn_kernel<<<blocksFor((long)B * 64 * T, TPB), TPB, 0, stream>>>(
        x, B, 1, T, ct_w, ct_b, 64, 3, 1,
        nullptr, nullptr, nullptr, nullptr, nullptr, 0, buf0);

    // ---- resblock 0: 64->64, identity skip (buf0 -> buf1) ----
    conv1d_bn_kernel<<<blocksFor((long)B * 64 * T, TPB), TPB, 0, stream>>>(
        buf0, B, 64, T, rb[0].c1w, rb[0].c1b, 64, 3, 1,
        rb[0].g1, rb[0].be1, rb[0].m1, rb[0].v1, nullptr, 1, buf2);
    conv1d_bn_kernel<<<blocksFor((long)B * 64 * T, TPB), TPB, 0, stream>>>(
        buf2, B, 64, T, rb[0].c2w, rb[0].c2b, 64, 3, 1,
        rb[0].g2, rb[0].be2, rb[0].m2, rb[0].v2, buf0, 1, buf1);

    // ---- resblock 1: 64->64 (buf1 -> buf0) ----
    conv1d_bn_kernel<<<blocksFor((long)B * 64 * T, TPB), TPB, 0, stream>>>(
        buf1, B, 64, T, rb[1].c1w, rb[1].c1b, 64, 3, 1,
        rb[1].g1, rb[1].be1, rb[1].m1, rb[1].v1, nullptr, 1, buf2);
    conv1d_bn_kernel<<<blocksFor((long)B * 64 * T, TPB), TPB, 0, stream>>>(
        buf2, B, 64, T, rb[1].c2w, rb[1].c2b, 64, 3, 1,
        rb[1].g2, rb[1].be2, rb[1].m2, rb[1].v2, buf1, 1, buf0);

    // ---- resblock 2: 64->128, conv skip (buf0 -> buf1) ----
    conv1d_bn_kernel<<<blocksFor((long)B * 128 * T, TPB), TPB, 0, stream>>>(
        buf0, B, 64, T, rb[2].skw, rb[2].skb, 128, 1, 0,
        nullptr, nullptr, nullptr, nullptr, nullptr, 0, skipb);
    conv1d_bn_kernel<<<blocksFor((long)B * 128 * T, TPB), TPB, 0, stream>>>(
        buf0, B, 64, T, rb[2].c1w, rb[2].c1b, 128, 3, 1,
        rb[2].g1, rb[2].be1, rb[2].m1, rb[2].v1, nullptr, 1, buf2);
    conv1d_bn_kernel<<<blocksFor((long)B * 128 * T, TPB), TPB, 0, stream>>>(
        buf2, B, 128, T, rb[2].c2w, rb[2].c2b, 128, 3, 1,
        rb[2].g2, rb[2].be2, rb[2].m2, rb[2].v2, skipb, 1, buf1);

    // ---- GAT layer 1 ----
    // A pack: A(m=node,k=chan) = buf1[b][k][m]   (B,512,128) f16
    pack_f16_kernel<<<blocksFor((long)B * Nn * 128, TPB), TPB, 0, stream>>>(
        buf1, 1, T, (long)128 * T, a1pack, Nn, 128, B);
    // BT pack: btW[n][k] = W1[k][n]   (64,128) f16
    pack_f16_kernel<<<blocksFor((long)64 * 128, TPB), TPB, 0, stream>>>(
        W1, 1, 64, 0, btW, 64, 128, 1);
    // Wh = A @ W1   (M=512, N=64, K=128)
    wmma_gemm_kernel<<<dim3(Nn / 64, 1, B), TPB, 0, stream>>>(
        a1pack, (long)Nn * 128, btW, 0, Whb, (long)Nn * 64, Nn, 64, 128);
    gat_scores_kernel<<<blocksFor((long)B * Nn, TPB), TPB, 0, stream>>>(
        Whb, a1, 64, B * Nn, f1b, f2b);
    attn_row_kernel<<<B * Nn, TPB, 0, stream>>>(f1b, f2b, attnh, Nn);
    // btWh[n][k] = Wh[k][n]   (B,64,512) f16
    pack_f16_kernel<<<blocksFor((long)B * 64 * Nn, TPB), TPB, 0, stream>>>(
        Whb, 1, 64, (long)Nn * 64, btWh, 64, Nn, B);
    // h' = attn @ Wh   (M=512, N=64, K=512)
    wmma_gemm_kernel<<<dim3(Nn / 64, 1, B), TPB, 0, stream>>>(
        attnh, (long)Nn * Nn, btWh, (long)64 * Nn, hprime, (long)Nn * 64, Nn, 64, Nn);
    elu_pack_kernel<<<blocksFor((long)B * Nn * 64, TPB), TPB, 0, stream>>>(
        hprime, hgath, nullptr, B * Nn * 64);

    // ---- GAT layer 2 ----
    pack_f16_kernel<<<blocksFor((long)64 * 64, TPB), TPB, 0, stream>>>(
        W2, 1, 64, 0, btW, 64, 64, 1);
    wmma_gemm_kernel<<<dim3(Nn / 64, 1, B), TPB, 0, stream>>>(
        hgath, (long)Nn * 64, btW, 0, Whb, (long)Nn * 64, Nn, 64, 64);
    gat_scores_kernel<<<blocksFor((long)B * Nn, TPB), TPB, 0, stream>>>(
        Whb, a2, 64, B * Nn, f1b, f2b);
    attn_row_kernel<<<B * Nn, TPB, 0, stream>>>(f1b, f2b, attnh, Nn);
    pack_f16_kernel<<<blocksFor((long)B * 64 * Nn, TPB), TPB, 0, stream>>>(
        Whb, 1, 64, (long)Nn * 64, btWh, 64, Nn, B);
    wmma_gemm_kernel<<<dim3(Nn / 64, 1, B), TPB, 0, stream>>>(
        attnh, (long)Nn * Nn, btWh, (long)64 * Nn, hprime, (long)Nn * 64, Nn, 64, Nn);
    elu_pack_kernel<<<blocksFor((long)B * Nn * 64, TPB), TPB, 0, stream>>>(
        hprime, hgath, hfin, B * Nn * 64);

    // ---- attention pool ----
    pool_logits_kernel<<<blocksFor((long)B * Nn, TPB), TPB, 0, stream>>>(
        hfin, pw1, pb1, pw2, pb2, logits, B * Nn, 64, 32);
    softmax_stats_kernel<<<B, TPB, 0, stream>>>(logits, stats, Nn);
    pool_out_kernel<<<B * 64, TPB, 0, stream>>>(
        hfin, logits, stats, (float*)d_out, Nn, 64);
}
